// YATE_Attention_34419867910594
// MI455X (gfx1250) — matrix operand order repaired and checked by reading, not compiled, not run on hardware
//
#include <hip/hip_runtime.h>
#include <hip/hip_bf16.h>
#include <math.h>

// ---------------------------------------------------------------------------
// YATE edge-attention for MI455X (gfx1250, wave32).
// GEMMs run as v_wmma_f32_16x16x32_bf16 (bf16 in, f32 accumulate): workload is
// ~190 FLOP/byte, far above what fp32 WMMA could sustain against 23.3 TB/s HBM,
// so the bf16 tensor path is the right precision/perf point.
// ---------------------------------------------------------------------------

#define N_NODES 20000
#define N_EDGES 320000
#define DIM     256
#define NHEADS  8

typedef __attribute__((ext_vector_type(16))) __bf16        v16bf;
typedef __attribute__((ext_vector_type(8)))  float         v8f;
typedef __attribute__((ext_vector_type(8)))  unsigned int  v8u;

// ws layout (float units)
#define WS_Q   0
#define WS_SC  (N_NODES * DIM)                 // E*H scores -> exp values
#define WS_M   (WS_SC + N_EDGES * NHEADS)      // N*H segment max
#define WS_D   (WS_M + N_NODES * NHEADS)       // N*H segment sum
#define WS_FLOATS (WS_D + N_NODES * NHEADS)    // = 8,000,000 floats (32 MB)
#define PK_ELEMS (DIM * DIM)                   // bf16 elems per packed matrix

__device__ __forceinline__ unsigned short f32_to_bf16_rne(float f) {
  unsigned int u = __float_as_uint(f);
  u += 0x7FFFu + ((u >> 16) & 1u);             // round-to-nearest-even
  return (unsigned short)(u >> 16);
}

// float atomic max via monotonic integer ordering (m initialized to -inf)
__device__ __forceinline__ void atomicMaxF32(float* addr, float val) {
  if (val >= 0.0f) atomicMax((int*)addr, __float_as_int(val));
  else             atomicMin((unsigned int*)addr, (unsigned int)__float_as_int(val));
}

// ---------------------------------------------------------------------------
// Kernel 0: pack weights to bf16 in WMMA B-fragment order; init m/denom/out.
// B-fragment order used here for V_WMMA_F32_16X16X32_BF16 (contiguous k, matching A):
//   lane = 16*h + n  (n = output col within 16-tile, h = K-half)
//   element j (0..15) of the lane's v16bf holds  B[K = 32*kt + 16*h + j][n]
// Packed address: ((nt*8 + kt)*32 + lane)*16 + j   -> one 32B load per lane.
// ---------------------------------------------------------------------------
__global__ void prep_kernel(const float* __restrict__ Wq, const float* __restrict__ Wk,
                            const float* __restrict__ Wv, const float* __restrict__ We,
                            float* __restrict__ ws_f, unsigned short* __restrict__ wpk,
                            float* __restrict__ out0) {
  int tid  = blockIdx.x * blockDim.x + threadIdx.x;
  int nthr = gridDim.x * blockDim.x;
  const float* mats[4] = {Wq, Wk, Wv, We};
  for (int m = 0; m < 4; ++m) {
    const float* W = mats[m];
    unsigned short* pk = wpk + m * PK_ELEMS;
    for (int t = tid; t < DIM * DIM; t += nthr) {
      int o = t >> 8, k = t & 255;             // W[o][k], used as B = W^T
      int nt = o >> 4, n = o & 15;
      int kt = k >> 5, kk = k & 31;
      int h = kk >> 4, j = kk & 15;
      pk[(((nt * 8 + kt) * 32) + (h * 16 + n)) * 16 + j] = f32_to_bf16_rne(W[t]);
    }
  }
  float* mbuf = ws_f + WS_M;
  float* dbuf = ws_f + WS_D;
  for (int t = tid; t < N_NODES * NHEADS; t += nthr) { mbuf[t] = -3.402823466e38f; dbuf[t] = 0.0f; }
  for (int t = tid; t < N_NODES * DIM; t += nthr) out0[t] = 0.0f;
}

// A-fragment gather from an LDS row-major bf16 tile [16][256].
// A order used here (contiguous k, as B is packed): lane row M = lane&15, h = lane>>4;
//   VGPR r holds K = 16*h + 2*r + {0,1}  (one b32 read).
__device__ __forceinline__ v16bf load_a_frag(const unsigned short* zt, int lane, int kt) {
  int mrow = lane & 15, hh = lane >> 4;
  v8u araw;
#pragma unroll
  for (int r = 0; r < 8; ++r) {
    int K = kt * 32 + hh * 16 + 2 * r;
    araw[r] = *(const unsigned int*)&zt[(mrow << 8) + K];
  }
  return __builtin_bit_cast(v16bf, araw);
}

__device__ __forceinline__ v16bf load_b_frag(const unsigned short* pk, int nt, int kt, int lane) {
  return __builtin_bit_cast(v16bf, *(const v8u*)&pk[(((nt * 8 + kt) * 32) + lane) * 16]);
}

// ---------------------------------------------------------------------------
// Kernel 1: q = x @ Wq^T   (1250 tiles of 16 rows; 8 waves x 2 col-tiles)
// ---------------------------------------------------------------------------
__global__ __launch_bounds__(256) void q_gemm_kernel(const float* __restrict__ x,
                                                     const unsigned short* __restrict__ wqpk,
                                                     float* __restrict__ qbuf) {
  __shared__ unsigned short xt[16 * DIM];
  int tid = threadIdx.x;
  int r0 = blockIdx.x * 16;
#pragma unroll 4
  for (int i = 0; i < 16; ++i)
    xt[i * DIM + tid] = f32_to_bf16_rne(x[(r0 + i) * DIM + tid]);
  __syncthreads();

  int w = tid >> 5, lane = tid & 31;
  int n = lane & 15, hh = lane >> 4;
  int nt0 = 2 * w, nt1 = 2 * w + 1;
  v8f acc0 = {}, acc1 = {};
  for (int kt = 0; kt < 8; ++kt) {
    v16bf a  = load_a_frag(xt, lane, kt);
    v16bf b0 = load_b_frag(wqpk, nt0, kt, lane);
    v16bf b1 = load_b_frag(wqpk, nt1, kt, lane);
    acc0 = __builtin_amdgcn_wmma_f32_16x16x32_bf16(false, a, false, b0, (short)0, acc0, false, false);
    acc1 = __builtin_amdgcn_wmma_f32_16x16x32_bf16(false, a, false, b1, (short)0, acc1, false, false);
  }
#pragma unroll
  for (int vr = 0; vr < 8; ++vr) {               // D: M = vr + 8*h, col = 16*nt + n
    int row = r0 + vr + 8 * hh;
    qbuf[row * DIM + nt0 * 16 + n] = acc0[vr];
    qbuf[row * DIM + nt1 * 16 + n] = acc1[vr];
  }
}

// ---------------------------------------------------------------------------
// Kernel 2: per 16-edge tile: Z = edge_attr * x[dst] (bf16, LDS);
//   k = Z@Wk^T, edge_out = Z@We^T + be; scores[e][h] = <q[src],k>_head/sqrt(32);
//   segment max via float atomics. Wave w's two col-tiles == head w.
// ---------------------------------------------------------------------------
__global__ __launch_bounds__(256) void edge_kq_kernel(
    const float* __restrict__ x, const int* __restrict__ ei,
    const float* __restrict__ ea, const float* __restrict__ be,
    const unsigned short* __restrict__ wkpk, const unsigned short* __restrict__ wepk,
    const float* __restrict__ qbuf,
    float* __restrict__ eout, float* __restrict__ scores, float* __restrict__ mbuf) {
  __shared__ unsigned short zt[16 * DIM];
  __shared__ int s_src[16];
  __shared__ int s_dst[16];
  int tid = threadIdx.x;
  int e0 = blockIdx.x * 16;
  if (tid < 16) { s_src[tid] = ei[e0 + tid]; s_dst[tid] = ei[N_EDGES + e0 + tid]; }
  __syncthreads();
#pragma unroll 4
  for (int i = 0; i < 16; ++i)
    zt[i * DIM + tid] = f32_to_bf16_rne(ea[(e0 + i) * DIM + tid] * x[s_dst[i] * DIM + tid]);
  __syncthreads();

  int w = tid >> 5, lane = tid & 31;
  int n = lane & 15, hh = lane >> 4;
  int nt0 = 2 * w, nt1 = 2 * w + 1;
  v8f k0 = {}, k1 = {}, eo0 = {}, eo1 = {};
  for (int kt = 0; kt < 8; ++kt) {
    v16bf a = load_a_frag(zt, lane, kt);
    v16bf bk0 = load_b_frag(wkpk, nt0, kt, lane);
    v16bf bk1 = load_b_frag(wkpk, nt1, kt, lane);
    v16bf bE0 = load_b_frag(wepk, nt0, kt, lane);
    v16bf bE1 = load_b_frag(wepk, nt1, kt, lane);
    k0  = __builtin_amdgcn_wmma_f32_16x16x32_bf16(false, a, false, bk0, (short)0, k0,  false, false);
    k1  = __builtin_amdgcn_wmma_f32_16x16x32_bf16(false, a, false, bk1, (short)0, k1,  false, false);
    eo0 = __builtin_amdgcn_wmma_f32_16x16x32_bf16(false, a, false, bE0, (short)0, eo0, false, false);
    eo1 = __builtin_amdgcn_wmma_f32_16x16x32_bf16(false, a, false, bE1, (short)0, eo1, false, false);
  }
  float bias0 = be[nt0 * 16 + n], bias1 = be[nt1 * 16 + n];
#pragma unroll
  for (int vr = 0; vr < 8; ++vr) {
    int erow = e0 + vr + 8 * hh;
    eout[(size_t)erow * DIM + nt0 * 16 + n] = eo0[vr] + bias0;
    eout[(size_t)erow * DIM + nt1 * 16 + n] = eo1[vr] + bias1;
  }
  // scores for head w: lane holds k cols {32w+n, 32w+16+n} for rows M=vr+8h
  float p[8];
#pragma unroll
  for (int vr = 0; vr < 8; ++vr) {
    int srcn = s_src[vr + 8 * hh];
    float q0 = qbuf[srcn * DIM + 32 * w + n];
    float q1 = qbuf[srcn * DIM + 32 * w + 16 + n];
    p[vr] = q0 * k0[vr] + q1 * k1[vr];
  }
#pragma unroll
  for (int vr = 0; vr < 8; ++vr)
#pragma unroll
    for (int off = 1; off < 16; off <<= 1)
      p[vr] += __shfl_xor(p[vr], off, 32);       // reduce across each 16-lane half
  if (n == 0) {
    const float inv_sqrt_c = 0.17677669529663687f;  // 1/sqrt(32)
#pragma unroll
    for (int vr = 0; vr < 8; ++vr) {
      int m_ = vr + 8 * hh;
      float s = p[vr] * inv_sqrt_c;
      scores[(e0 + m_) * NHEADS + w] = s;
      atomicMaxF32(&mbuf[s_src[m_] * NHEADS + w], s);
    }
  }
}

// ---------------------------------------------------------------------------
// Kernel 3: ex = exp(score - m[src]); denom[src] += ex  (elementwise E*H)
// ---------------------------------------------------------------------------
__global__ void softmax_kernel(const int* __restrict__ ei, const float* __restrict__ mbuf,
                               float* __restrict__ scores, float* __restrict__ dbuf) {
  int idx = blockIdx.x * blockDim.x + threadIdx.x;
  if (idx >= N_EDGES * NHEADS) return;
  int e = idx >> 3, h = idx & 7;
  int s = ei[e];
  float ex = __expf(scores[idx] - mbuf[s * NHEADS + h]);
  scores[idx] = ex;
  atomicAdd(&dbuf[s * NHEADS + h], ex);
}

// ---------------------------------------------------------------------------
// Kernel 4: recompute Z, v = Z@Wv^T; out[src] += (ex/denom) * v
// (recompute avoids a 327MB v spill to scratch of unknown size)
// ---------------------------------------------------------------------------
__global__ __launch_bounds__(256) void edge_v_kernel(
    const float* __restrict__ x, const int* __restrict__ ei,
    const float* __restrict__ ea, const unsigned short* __restrict__ wvpk,
    const float* __restrict__ exbuf, const float* __restrict__ dbuf,
    float* __restrict__ out) {
  __shared__ unsigned short zt[16 * DIM];
  __shared__ int s_src[16];
  __shared__ int s_dst[16];
  int tid = threadIdx.x;
  int e0 = blockIdx.x * 16;
  if (tid < 16) { s_src[tid] = ei[e0 + tid]; s_dst[tid] = ei[N_EDGES + e0 + tid]; }
  __syncthreads();
#pragma unroll 4
  for (int i = 0; i < 16; ++i)
    zt[i * DIM + tid] = f32_to_bf16_rne(ea[(e0 + i) * DIM + tid] * x[s_dst[i] * DIM + tid]);
  __syncthreads();

  int w = tid >> 5, lane = tid & 31;
  int n = lane & 15, hh = lane >> 4;
  int nt0 = 2 * w, nt1 = 2 * w + 1;
  v8f v0 = {}, v1 = {};
  for (int kt = 0; kt < 8; ++kt) {
    v16bf a  = load_a_frag(zt, lane, kt);
    v16bf b0 = load_b_frag(wvpk, nt0, kt, lane);
    v16bf b1 = load_b_frag(wvpk, nt1, kt, lane);
    v0 = __builtin_amdgcn_wmma_f32_16x16x32_bf16(false, a, false, b0, (short)0, v0, false, false);
    v1 = __builtin_amdgcn_wmma_f32_16x16x32_bf16(false, a, false, b1, (short)0, v1, false, false);
  }
#pragma unroll
  for (int vr = 0; vr < 8; ++vr) {
    int m_ = vr + 8 * hh;
    int srcn = s_src[m_];
    float attn = exbuf[(e0 + m_) * NHEADS + w] / (dbuf[srcn * NHEADS + w] + 1e-16f);
    atomicAdd(&out[srcn * DIM + 32 * w + n],      attn * v0[vr]);
    atomicAdd(&out[srcn * DIM + 32 * w + 16 + n], attn * v1[vr]);
  }
}

// ---------------------------------------------------------------------------
extern "C" void kernel_launch(void* const* d_in, const int* in_sizes, int n_in,
                              void* d_out, int out_size, void* d_ws, size_t ws_size,
                              hipStream_t stream) {
  const float* x  = (const float*)d_in[0];
  const int*   ei = (const int*)d_in[1];       // [2, E]: row0 = src, row1 = dst
  const float* ea = (const float*)d_in[2];
  const float* Wq = (const float*)d_in[3];
  const float* Wk = (const float*)d_in[4];
  const float* Wv = (const float*)d_in[5];
  const float* We = (const float*)d_in[6];
  const float* be = (const float*)d_in[7];

  float* out  = (float*)d_out;                           // [N, DIM]
  float* eout = out + (size_t)N_NODES * DIM;             // [E, DIM]

  float* wsf = (float*)d_ws;
  float* qbuf   = wsf + WS_Q;
  float* scores = wsf + WS_SC;
  float* mbuf   = wsf + WS_M;
  float* dbuf   = wsf + WS_D;
  unsigned short* wpk = (unsigned short*)((char*)d_ws + (size_t)WS_FLOATS * 4);

  prep_kernel<<<2048, 256, 0, stream>>>(Wq, Wk, Wv, We, wsf, wpk, out);
  q_gemm_kernel<<<N_NODES / 16, 256, 0, stream>>>(x, wpk + 0 * PK_ELEMS, qbuf);
  edge_kq_kernel<<<N_EDGES / 16, 256, 0, stream>>>(x, ei, ea, be,
                                                   wpk + 1 * PK_ELEMS, wpk + 3 * PK_ELEMS,
                                                   qbuf, eout, scores, mbuf);
  softmax_kernel<<<(N_EDGES * NHEADS + 255) / 256, 256, 0, stream>>>(ei, mbuf, scores, dbuf);
  edge_v_kernel<<<N_EDGES / 16, 256, 0, stream>>>(x, ei, ea, wpk + 2 * PK_ELEMS,
                                                  scores, dbuf, out);
}
